// LipFormerDecoder_83588653514953
// MI455X (gfx1250) — matrix operand order repaired and checked by reading, not compiled, not run on hardware
//
#include <hip/hip_runtime.h>

// ---------------------------------------------------------------------------
// LipFormer encoder-decoder GRU + attention for MI455X (gfx1250, wave32).
// GEMMs via v_wmma_f32_16x16x32_bf16; A-tiles DMA'd into LDS by the Tensor
// Data Mover (tensor_load_to_lds, double-buffered, s_wait_tensorcnt), B-tiles
// staged manually (transposed). Weights + enc_proj + h_p_e kept bf16 so the
// decoder's 75 sweeps over them stay resident in the 192 MB L2.
// ---------------------------------------------------------------------------

#define BB 512
#define TT 75
#define DD 1024
#define HH 1024
#define VV 64

typedef __attribute__((ext_vector_type(16))) __bf16          v16bf;
typedef __attribute__((ext_vector_type(16))) unsigned short  v16us;
typedef __attribute__((ext_vector_type(8)))  unsigned short  u16x8;
typedef __attribute__((ext_vector_type(8)))  float           v8f;
typedef __attribute__((ext_vector_type(4)))  unsigned int    v4u;
typedef __attribute__((ext_vector_type(8)))  int             v8i;
typedef __attribute__((ext_vector_type(4)))  int             v4i;

__device__ __forceinline__ unsigned short f32_to_bf16(float f) {
  unsigned int u = __float_as_uint(f);
  unsigned int r = 0x7FFFu + ((u >> 16) & 1u);   // round-to-nearest-even
  return (unsigned short)((u + r) >> 16);
}
__device__ __forceinline__ float bf16_to_f32(unsigned short h) {
  return __uint_as_float(((unsigned int)h) << 16);
}

// ---------------------------------------------------------------------------
// GEMM: C[M,N](f32 or bf16) = A[M,K](bf16,row-major,lda) @ B[K,N](bf16,ldb)
//       (+ bias[N] if bias != nullptr)
// Block tile 128x64, 8 waves, each wave a 32x32 tile = 2x2 WMMA 16x16.
// Requires: M%128==0, N%64==0, K%32==0 (true at every call site here).
// A tile: TDM tensor_load_to_lds (double-buffered; DMA of tile k+1 overlaps
//         WMMA on tile k). B tile: manual transpose into LDS.
// ---------------------------------------------------------------------------
#define TILE_M 128
#define TILE_N 64
#define TILE_K 32
#define LDS_S  40   // 32 + 8 pad: 64B row + 16B pad (TDM: interval 16 DW, amount 4 DW)

#if __has_builtin(__builtin_amdgcn_tensor_load_to_lds)
#define USE_TDM 1
#else
#define USE_TDM 0
#endif

#if USE_TDM
// Issue one TDM load: 128 rows x 32 cols of bf16 from A(global, stride lda
// elements) into LDS at lds_byte_off with 16B padding after each 64B row.
// This toolchain exposes the 6-arg builtin:
//   (uint32x4 g0, int32x8 g1, int32x4 g2, int32x4 g3, int32x8 g4, i32 cpol)
__device__ __forceinline__ void tdm_load_A(const unsigned short* gptr,
                                           unsigned lds_byte_off, int lda) {
  unsigned long long ga = (unsigned long long)(const void*)gptr;
  v4u g0;
  g0[0] = 1u;                                   // count=1 (valid descriptor)
  g0[1] = lds_byte_off;                         // D#.lds_addr
  g0[2] = (unsigned)ga;                         // D#.global_addr[31:0]
  g0[3] = (unsigned)((ga >> 32) & 0x1FFFFFFu)   // D#.global_addr[56:32]
        | (2u << 30);                           // D#.type = 2 ("image")
  v8i g1;
  g1[0] = (1 << 16)                             // data_size = 2 bytes
        | (1 << 20)                             // pad_enable
        | (3 << 22)                             // pad_interval: 16 DWORDs (64B row)
        | (3 << 25);                            // pad_amount: 4 DWORDs (16B)
  g1[1] = (TILE_K << 16);                       // tensor_dim0 = 32 (lo16)
  g1[2] = (TILE_M << 16);                       // tensor_dim0 hi | tensor_dim1 lo = 128
  g1[3] = (TILE_K << 16);                       // tensor_dim1 hi | tile_dim0 = 32
  g1[4] = TILE_M;                               // tile_dim1 = 128 | tile_dim2 = 0
  g1[5] = lda;                                  // tensor_dim0_stride lo32 (elements)
  g1[6] = 0;                                    // stride hi | tensor_dim1_stride lo
  g1[7] = 0;
  v4i g2 = {0, 0, 0, 0};                        // D# group 2 (unused, <=2D tensor)
  v4i g3 = {0, 0, 0, 0};                        // D# group 3 (unused)
  v8i g4 = {0, 0, 0, 0, 0, 0, 0, 0};           // toolchain-specific extra group
  __builtin_amdgcn_tensor_load_to_lds(g0, g1, g2, g3, g4, 0);
}
#endif

template <bool STORE_BF16>
__global__ __launch_bounds__(256) void gemm_bf16_wmma(
    const unsigned short* __restrict__ A, int lda,
    const unsigned short* __restrict__ B, int ldb,
    void* __restrict__ Cout, int ldc,
    const float* __restrict__ bias,
    int M, int N, int K)
{
  (void)M; (void)N;
  __shared__ unsigned short lA[2][TILE_M * LDS_S];  // [buf][m][k] (double-buffered)
  __shared__ unsigned short lB[TILE_N * LDS_S];     // transposed: [n][k]

  const int tid  = threadIdx.x;
  const int lane = tid & 31;
  const int wid  = tid >> 5;            // 0..7
  const int wm   = (wid & 3) * 32;      // wave M offset inside tile
  const int wn   = (wid >> 2) * 32;     // wave N offset inside tile
  const int hh   = lane >> 4;           // lane half (ISA A/B layout split)
  const int l16  = lane & 15;

  const int m0 = blockIdx.y * TILE_M;
  const int n0 = blockIdx.x * TILE_N;

  v8f acc[2][2] = {};

#if USE_TDM
  if (wid == 0)  // one wave per block drives the Tensor Data Mover
    tdm_load_A(A + (size_t)m0 * lda, (unsigned)(size_t)&lA[0][0], lda);
#endif

  int cur = 0;
  for (int kb = 0; kb < K; kb += TILE_K, cur ^= 1) {
#if !USE_TDM
    // manual A staging: 128x32, two 128-bit transfers per thread
    #pragma unroll
    for (int i = 0; i < 2; ++i) {
      int idx = tid + i * 256;          // 0..511
      int r = idx >> 2, c8 = (idx & 3) * 8;
      *(u16x8*)&lA[cur][r * LDS_S + c8] =
          *(const u16x8*)(A + (size_t)(m0 + r) * lda + kb + c8);
    }
#endif
    // B staging: 32x64 loaded 128-bit coalesced, stored transposed [n][k]
    {
      int k = tid >> 3, n8 = (tid & 7) * 8;
      u16x8 bv = *(const u16x8*)(B + (size_t)(kb + k) * ldb + n0 + n8);
      #pragma unroll
      for (int j = 0; j < 8; ++j) lB[(n8 + j) * LDS_S + k] = bv[j];
      int kpf = (kb + TILE_K < K) ? (kb + TILE_K) : kb;  // branch-free clamp
      __builtin_prefetch(B + (size_t)(kpf + k) * ldb + n0 + n8, 0, 1);
    }
#if USE_TDM
    if (wid == 0) __builtin_amdgcn_s_wait_tensorcnt(0);  // lA[cur] DMA done
#endif
    __syncthreads();
#if USE_TDM
    if (wid == 0 && kb + TILE_K < K)    // kick DMA for next tile into lA[cur^1]
      tdm_load_A(A + (size_t)m0 * lda + kb + TILE_K,
                 (unsigned)(size_t)&lA[cur ^ 1][0], lda);
#endif

    // Fragments per ISA 7.12.2:
    //  A 16x32 bf16: lane half 0 holds K {0..7,16..23}, half 1 {8..15,24..31}
    //  B 32x16 bf16: same K order as A: half 0 K {0..7,16..23}, half 1 {8..15,24..31}
    v16bf af[2], bfr[2];
    #pragma unroll
    for (int mi = 0; mi < 2; ++mi) {
      v16us au;
      const unsigned short* pa = &lA[cur][(wm + mi * 16 + l16) * LDS_S];
      #pragma unroll
      for (int j = 0; j < 8; ++j) {
        int kk = ((j < 4) ? (2 * j) : (16 + 2 * (j - 4))) + 8 * hh;
        au[2 * j]     = pa[kk];
        au[2 * j + 1] = pa[kk + 1];
      }
      af[mi] = __builtin_bit_cast(v16bf, au);
    }
    #pragma unroll
    for (int ni = 0; ni < 2; ++ni) {
      v16us bu;
      const unsigned short* pb = &lB[(wn + ni * 16 + l16) * LDS_S];
      #pragma unroll
      for (int j = 0; j < 8; ++j) {
        int kk = ((j < 4) ? (2 * j) : (16 + 2 * (j - 4))) + 8 * hh;
        bu[2 * j]     = pb[kk];
        bu[2 * j + 1] = pb[kk + 1];
      }
      bfr[ni] = __builtin_bit_cast(v16bf, bu);
    }
    #pragma unroll
    for (int mi = 0; mi < 2; ++mi)
      #pragma unroll
      for (int ni = 0; ni < 2; ++ni)
        acc[mi][ni] = __builtin_amdgcn_wmma_f32_16x16x32_bf16(
            false, af[mi], false, bfr[ni], (short)0, acc[mi][ni], false, false);
    __syncthreads();
  }

  // Epilogue. C layout: VGPR r, lane l -> M = r + 8*(l>=16), N = l&15.
  #pragma unroll
  for (int mi = 0; mi < 2; ++mi) {
    #pragma unroll
    for (int ni = 0; ni < 2; ++ni) {
      int col = n0 + wn + ni * 16 + l16;
      float bv = bias ? bias[col] : 0.0f;
      #pragma unroll
      for (int r = 0; r < 8; ++r) {
        int row = m0 + wm + mi * 16 + r + 8 * hh;
        float v = acc[mi][ni][r] + bv;
        if (STORE_BF16)
          ((unsigned short*)Cout)[(size_t)row * ldc + col] = f32_to_bf16(v);
        else
          ((float*)Cout)[(size_t)row * ldc + col] = v;
      }
    }
  }
}

// ---------------------------------------------------------------------------
// Elementwise / reduction kernels
// ---------------------------------------------------------------------------
__global__ void k_cvt_bf16(const float* __restrict__ src,
                           unsigned short* __restrict__ dst, int n) {
  int i = blockIdx.x * 256 + threadIdx.x;
  if (i < n) dst[i] = f32_to_bf16(src[i]);
}

__global__ void k_embsum(const float* __restrict__ emb, float* __restrict__ es) {
  int v = threadIdx.x;  // 64
  float s = 0.0f;
  for (int h = 0; h < HH; ++h) s += emb[h * VV + v];
  es[v] = s;
}

// Keras GRU (reset_after=true). xproj/inner are [B,3H] with per-side biases
// already folded in by the GEMM. In-place f32 state update + bf16 copies.
__global__ void k_gru(const float* __restrict__ xproj,
                      const float* __restrict__ inner,
                      float* __restrict__ hio,
                      unsigned short* __restrict__ outA, long long strideA,
                      unsigned short* __restrict__ outB, long long strideB) {
  int idx = blockIdx.x * 256 + threadIdx.x;  // B*H threads
  int b = idx >> 10, k = idx & 1023;
  const float* xp = xproj + (size_t)b * (3 * HH);
  const float* ip = inner + (size_t)b * (3 * HH);
  float xz = xp[k], xr = xp[k + HH], xh = xp[k + 2 * HH];
  float hz = ip[k], hr = ip[k + HH], hh = ip[k + 2 * HH];
  float hp = hio[idx];
  float z = 1.0f / (1.0f + __expf(-(xz + hz)));
  float r = 1.0f / (1.0f + __expf(-(xr + hr)));
  float c = tanhf(xh + r * hh);
  float hn = z * hp + (1.0f - z) * c;
  hio[idx] = hn;
  unsigned short hb = f32_to_bf16(hn);
  if (outA) outA[(size_t)b * strideA + k] = hb;
  if (outB) outB[(size_t)b * strideB + k] = hb;
}

// scores[b,t] = b2 + sum_k W2[k] * tanh(dproj[b,k] + enc_proj[b,t,k])
// (fuses att_h: never materializes the 512x75x1024 tensor)
__global__ void k_scores(const float* __restrict__ dproj,
                         const unsigned short* __restrict__ encproj,
                         const float* __restrict__ W2,
                         const float* __restrict__ b2,
                         float* __restrict__ scores) {
  int t = blockIdx.x, b = blockIdx.y, tid = threadIdx.x;  // 256 threads
  const float* dp = dproj + (size_t)b * HH;
  const unsigned short* ep = encproj + ((size_t)b * TT + t) * HH;
  float partial = 0.0f;
  #pragma unroll
  for (int k = tid; k < HH; k += 256)
    partial += W2[k] * tanhf(dp[k] + bf16_to_f32(ep[k]));
  __shared__ float red[256];
  red[tid] = partial;
  __syncthreads();
  for (int s = 128; s > 0; s >>= 1) {
    if (tid < s) red[tid] += red[tid + s];
    __syncthreads();
  }
  if (tid == 0) scores[(size_t)b * TT + t] = red[0] + b2[0];
}

__global__ void k_softmax_att(const float* __restrict__ scores,
                              float* __restrict__ att) {
  int b = blockIdx.x, tid = threadIdx.x;  // 128 threads, 75 active
  __shared__ float red[128];
  float v = (tid < TT) ? scores[(size_t)b * TT + tid] : -3.4e38f;
  red[tid] = v;
  __syncthreads();
  for (int s = 64; s > 0; s >>= 1) {
    if (tid < s) red[tid] = fmaxf(red[tid], red[tid + s]);
    __syncthreads();
  }
  float mx = red[0];
  __syncthreads();
  float e = (tid < TT) ? __expf(v - mx) : 0.0f;
  red[tid] = e;
  __syncthreads();
  for (int s = 64; s > 0; s >>= 1) {
    if (tid < s) red[tid] += red[tid + s];
    __syncthreads();
  }
  if (tid < TT) att[(size_t)b * TT + tid] = e / red[0];
}

// context[b,k] = sum_t att[b,t] * h_p_e[b,t,k]  -> bf16 into concat[:,H+k]
__global__ void k_context(const float* __restrict__ att,
                          const unsigned short* __restrict__ hpe,
                          unsigned short* __restrict__ concat) {
  int idx = blockIdx.x * 256 + threadIdx.x;
  int b = idx >> 10, k = idx & 1023;
  const float* ab = att + (size_t)b * TT;
  const unsigned short* hb = hpe + (size_t)b * TT * HH + k;
  float acc = 0.0f;
  #pragma unroll 5
  for (int t = 0; t < TT; ++t) acc += ab[t] * bf16_to_f32(hb[(size_t)t * HH]);
  concat[(size_t)b * (2 * HH) + HH + k] = f32_to_bf16(acc);
}

// softmax over logits -> write p into d_out[:,t,:]; emb_out = p * emb_sum
__global__ void k_softmax_out(const float* __restrict__ logits,
                              const float* __restrict__ embsum,
                              float* __restrict__ out,
                              unsigned short* __restrict__ embout, int t) {
  int b = blockIdx.x, v = threadIdx.x;  // 64 threads
  __shared__ float red[64];
  float x = logits[(size_t)b * VV + v];
  red[v] = x;
  __syncthreads();
  for (int s = 32; s > 0; s >>= 1) {
    if (v < s) red[v] = fmaxf(red[v], red[v + s]);
    __syncthreads();
  }
  float mx = red[0];
  __syncthreads();
  float e = __expf(x - mx);
  red[v] = e;
  __syncthreads();
  for (int s = 32; s > 0; s >>= 1) {
    if (v < s) red[v] += red[v + s];
    __syncthreads();
  }
  float p = e / red[0];
  out[((size_t)b * TT + t) * VV + v] = p;
  embout[(size_t)b * VV + v] = f32_to_bf16(p * embsum[v]);
}

// ---------------------------------------------------------------------------
// Host orchestration
// ---------------------------------------------------------------------------
static inline void launch_gemm(const unsigned short* A, int lda,
                               const unsigned short* B, int ldb,
                               void* C, int ldc, const float* bias,
                               int M, int N, int K, bool storeBf16,
                               hipStream_t s) {
  dim3 grid(N / TILE_N, M / TILE_M), block(256);
  if (storeBf16)
    gemm_bf16_wmma<true><<<grid, block, 0, s>>>(A, lda, B, ldb, C, ldc, bias, M, N, K);
  else
    gemm_bf16_wmma<false><<<grid, block, 0, s>>>(A, lda, B, ldb, C, ldc, bias, M, N, K);
}

extern "C" void kernel_launch(void* const* d_in, const int* in_sizes, int n_in,
                              void* d_out, int out_size, void* d_ws, size_t ws_size,
                              hipStream_t stream) {
  (void)in_sizes; (void)n_in; (void)out_size; (void)ws_size;
  const float* inputs      = (const float*)d_in[0];
  const float* enc_kernel  = (const float*)d_in[1];
  const float* enc_rkernel = (const float*)d_in[2];
  const float* enc_bias    = (const float*)d_in[3];   // [2,3H]
  const float* emb         = (const float*)d_in[4];
  const float* dec_kernel  = (const float*)d_in[5];
  const float* dec_rkernel = (const float*)d_in[6];
  const float* dec_bias    = (const float*)d_in[7];   // [2,3H]
  const float* W1          = (const float*)d_in[8];   // [2H,H] (dec rows then enc rows)
  const float* b1          = (const float*)d_in[9];
  const float* W2          = (const float*)d_in[10];  // [H,1]
  const float* b2          = (const float*)d_in[11];
  const float* W3          = (const float*)d_in[12];  // [2H,V]
  const float* b3          = (const float*)d_in[13];

  // ---- workspace carve (total ~285 MB) ----
  char* ws = (char*)d_ws;
  size_t off = 0;
  auto carve = [&](size_t bytes) -> void* {
    off = (off + 255) & ~(size_t)255;
    void* p = ws + off;
    off += bytes;
    return p;
  };
  unsigned short* inputs_bf = (unsigned short*)carve((size_t)BB * TT * DD * 2);
  unsigned short* wEncK     = (unsigned short*)carve((size_t)DD * 3 * HH * 2);
  unsigned short* wEncR     = (unsigned short*)carve((size_t)HH * 3 * HH * 2);
  unsigned short* wDecK     = (unsigned short*)carve((size_t)VV * 3 * HH * 2);
  unsigned short* wDecR     = (unsigned short*)carve((size_t)HH * 3 * HH * 2);
  unsigned short* wW1       = (unsigned short*)carve((size_t)2 * HH * HH * 2);
  unsigned short* wW3       = (unsigned short*)carve((size_t)2 * HH * VV * 2);
  float*          embsum    = (float*)carve(VV * 4);
  unsigned short* hpe_bf    = (unsigned short*)carve((size_t)BB * TT * HH * 2);
  unsigned short* encproj   = (unsigned short*)carve((size_t)BB * TT * HH * 2);
  float*          xproj     = (float*)carve((size_t)BB * 3 * HH * 4);
  float*          inner     = (float*)carve((size_t)BB * 3 * HH * 4);
  float*          henc      = (float*)carve((size_t)BB * HH * 4);
  unsigned short* henc_bf   = (unsigned short*)carve((size_t)BB * HH * 2);
  float*          hdec      = (float*)carve((size_t)BB * HH * 4);
  unsigned short* concat    = (unsigned short*)carve((size_t)BB * 2 * HH * 2);  // [hd | ctx]
  float*          dproj     = (float*)carve((size_t)BB * HH * 4);
  float*          scores    = (float*)carve((size_t)BB * TT * 4);
  float*          att       = (float*)carve((size_t)BB * TT * 4);
  float*          logits    = (float*)carve((size_t)BB * VV * 4);
  unsigned short* embout    = (unsigned short*)carve((size_t)BB * VV * 2);

  auto cvt = [&](const float* s, unsigned short* d, int n) {
    k_cvt_bf16<<<(n + 255) / 256, 256, 0, stream>>>(s, d, n);
  };
  // ---- one-time (per launch) prep: bf16 weight conversion, emb_sum ----
  cvt(inputs, inputs_bf, BB * TT * DD);
  cvt(enc_kernel, wEncK, DD * 3 * HH);
  cvt(enc_rkernel, wEncR, HH * 3 * HH);
  cvt(dec_kernel, wDecK, VV * 3 * HH);
  cvt(dec_rkernel, wDecR, HH * 3 * HH);
  cvt(W1, wW1, 2 * HH * HH);
  cvt(W3, wW3, 2 * HH * VV);
  k_embsum<<<1, VV, 0, stream>>>(emb, embsum);

  (void)hipMemsetAsync(henc, 0, (size_t)BB * HH * 4, stream);
  (void)hipMemsetAsync(henc_bf, 0, (size_t)BB * HH * 2, stream);
  (void)hipMemsetAsync(hdec, 0, (size_t)BB * HH * 4, stream);
  (void)hipMemsetAsync(concat, 0, (size_t)BB * 2 * HH * 2, stream);
  (void)hipMemsetAsync(embout, 0, (size_t)BB * VV * 2, stream);

  const int gruGrid = BB * HH / 256;

  // ---- encoder recurrence ----
  for (int t = 0; t < TT; ++t) {
    // x_proj_t = inputs[:,t,:] @ enc_kernel + enc_bias[0]
    launch_gemm(inputs_bf + (size_t)t * DD, TT * DD, wEncK, 3 * HH,
                xproj, 3 * HH, enc_bias, BB, 3 * HH, DD, false, stream);
    // inner = h @ enc_rkernel + enc_bias[1]
    launch_gemm(henc_bf, HH, wEncR, 3 * HH,
                inner, 3 * HH, enc_bias + 3 * HH, BB, 3 * HH, HH, false, stream);
    k_gru<<<gruGrid, 256, 0, stream>>>(xproj, inner, henc,
                                       henc_bf, HH,
                                       hpe_bf + (size_t)t * HH, (long long)TT * HH);
  }

  // enc_proj = h_p_e @ W1_enc + b1   (stored bf16 -> L2-resident for decoder)
  launch_gemm(hpe_bf, HH, wW1 + (size_t)HH * HH, HH,
              encproj, HH, b1, BB * TT, HH, HH, true, stream);

  // ---- decoder recurrence ----
  for (int t = 0; t < TT; ++t) {
    // x = emb_out @ dec_kernel + dec_bias[0]
    launch_gemm(embout, VV, wDecK, 3 * HH,
                xproj, 3 * HH, dec_bias, BB, 3 * HH, VV, false, stream);
    // inner = hd @ dec_rkernel + dec_bias[1]  (hd bf16 lives in concat[:, :H])
    launch_gemm(concat, 2 * HH, wDecR, 3 * HH,
                inner, 3 * HH, dec_bias + 3 * HH, BB, 3 * HH, HH, false, stream);
    k_gru<<<gruGrid, 256, 0, stream>>>(xproj, inner, hdec,
                                       concat, 2 * HH, nullptr, 0);
    // dproj = hd @ W1_dec (bias is inside enc_proj)
    launch_gemm(concat, 2 * HH, wW1, HH,
                dproj, HH, nullptr, BB, HH, HH, false, stream);
    k_scores<<<dim3(TT, BB), 256, 0, stream>>>(dproj, encproj, W2, b2, scores);
    k_softmax_att<<<BB, 128, 0, stream>>>(scores, att);
    k_context<<<gruGrid, 256, 0, stream>>>(att, hpe_bf, concat);
    // logits = [hd | context] @ W3 + b3
    launch_gemm(concat, 2 * HH, wW3, VV,
                logits, VV, b3, BB, VV, 2 * HH, false, stream);
    k_softmax_out<<<BB, VV, 0, stream>>>(logits, embsum, (float*)d_out, embout, t);
  }
}
